// AttentionModule_17927193493644
// MI455X (gfx1250) — compile-verified, matrix operand order checked by reading, not run on hardware
//
#include <hip/hip_runtime.h>

// ---------------------------------------------------------------------------
// Windowed multi-head self-attention (Swin-style) for MI455X / gfx1250.
// B=1024 windows, N=64 tokens, C=512, H=16 heads, D=32. All WMMA bf16.
// Double-buffered GLOBAL_LOAD_ASYNC_TO_LDS_B128 staging (ASYNCcnt).
// ---------------------------------------------------------------------------

typedef __attribute__((ext_vector_type(16))) __bf16 v16bf;
typedef __attribute__((ext_vector_type(8)))  float  v8f;
typedef __attribute__((ext_vector_type(4)))  int    v4i;

union frag_u { v16bf v; uint4 q[2]; };

// ---- async global->LDS copy (gfx1250), guarded fallbacks ------------------
#if defined(__has_builtin)
#  if __has_builtin(__builtin_amdgcn_global_load_async_to_lds_b128)
#    define HAVE_ASYNC_LDS 1
#  else
#    define HAVE_ASYNC_LDS 0
#  endif
#  if __has_builtin(__builtin_amdgcn_s_wait_asynccnt)
#    define HAVE_WAIT_ASYNC 1
#  else
#    define HAVE_WAIT_ASYNC 0
#  endif
#else
#  define HAVE_ASYNC_LDS 0
#  define HAVE_WAIT_ASYNC 0
#endif

typedef __attribute__((address_space(1))) v4i* as1_v4i;
typedef __attribute__((address_space(3))) v4i* as3_v4i;

__device__ __forceinline__ as3_v4i to_lds(void* p) {
    // LDS offset = low 32 bits of the generic (aperture) address
    return (as3_v4i)(unsigned int)(unsigned long long)p;
}
__device__ __forceinline__ as1_v4i to_glb(const void* p) {
    return (as1_v4i)(unsigned long long)p;
}

__device__ __forceinline__ void async_copy_b128(const void* g, void* l) {
#if HAVE_ASYNC_LDS
    __builtin_amdgcn_global_load_async_to_lds_b128(to_glb(g), to_lds(l), 0, 0);
#else
    *(uint4*)l = *(const uint4*)g;
#endif
}

__device__ __forceinline__ void async_join() {
#if HAVE_ASYNC_LDS
#  if HAVE_WAIT_ASYNC
    __builtin_amdgcn_s_wait_asynccnt(0);
#  else
    asm volatile("s_wait_asynccnt 0" ::: "memory");
#  endif
#endif
}

// ---------------------------------------------------------------------------
__device__ __forceinline__ unsigned short f2bf(float f) {
    unsigned int u = __float_as_uint(f);
    unsigned int r = (u + 0x7FFFu + ((u >> 16) & 1u)) >> 16;   // RNE
    return (unsigned short)r;
}

// Fragment load: 16 rows x 32 K from row-major bf16 (ld elements). Per ISA
// 7.12.2 each half-register-file is 8 contiguous K values => 2x ds_load_b128.
__device__ __forceinline__ v16bf frag_ld(const unsigned short* src, int row0, int k0, int ld) {
    const int lane = threadIdx.x & 31;
    const int kh   = (lane >> 4) * 8;          // 0 or 8
    const unsigned short* p = src + (row0 + (lane & 15)) * ld + k0 + kh;
    frag_u f;
    f.q[0] = *(const uint4*)(p);               // K = kh .. kh+7
    f.q[1] = *(const uint4*)(p + 16);          // K = 16+kh .. 16+kh+7
    return f.v;
}

__device__ __forceinline__ v8f wmma_bf16(v16bf a, v16bf b, v8f c) {
    return __builtin_amdgcn_wmma_f32_16x16x32_bf16(false, a, false, b, (short)0, c, false, false);
}

// ---------------------------------------------------------------------------
// Kernel 1a: f32 -> bf16 packing (x)
// ---------------------------------------------------------------------------
__global__ void cvt_bf16(const float* __restrict__ in, unsigned short* __restrict__ out, int n) {
    int i = blockIdx.x * 256 + threadIdx.x;
    if (i < n) out[i] = f2bf(in[i]);
}

// Kernel 1b: f32 -> bf16 with transpose (weights): out[c*rows + k] = in[k*cols + c]
__global__ void cvtT_bf16(const float* __restrict__ in, unsigned short* __restrict__ out,
                          int rows, int cols) {
    int i = blockIdx.x * 256 + threadIdx.x;
    if (i < rows * cols) {
        int k = i / cols, c = i % cols;
        out[c * rows + k] = f2bf(in[i]);
    }
}

// ---------------------------------------------------------------------------
// Kernel 2: fused QKV projection GEMM  [65536,512] x [512,1536] (WqT | WkvT)
// Block tile 128x64, 8 waves, each wave 32x32 (4 WMMAs / K-step).
// Double-buffered async LDS staging; one barrier per K-step.
// ---------------------------------------------------------------------------
__global__ __launch_bounds__(256)
void qkv_gemm(const unsigned short* __restrict__ xb,
              const unsigned short* __restrict__ wqt,   // [512 out][512 in]
              const unsigned short* __restrict__ wkvt,  // [1024 out][512 in]
              const float* __restrict__ bq,
              const float* __restrict__ bkv,
              unsigned short* __restrict__ Qb,
              unsigned short* __restrict__ Kb,
              unsigned short* __restrict__ Vb)
{
    __shared__ __attribute__((aligned(16))) unsigned short As[2][128 * 32];
    __shared__ __attribute__((aligned(16))) unsigned short Bt[2][64 * 32];

    const int tid  = threadIdx.x;
    const int wave = tid >> 5;
    const int row0 = blockIdx.y * 128;
    const int col0 = blockIdx.x * 64;            // 0..1535 over concat(Wq|Wkv)

    const int wm = (wave & 3) * 32;
    const int wn = (wave >> 2) * 32;

    // hoisted per-thread staging pointers
    const int r0 = tid >> 2, c4 = tid & 3;
    const unsigned short* aG0 = xb + (row0 + r0) * 512 + c4 * 8;
    const unsigned short* aG1 = aG0 + 64 * 512;
    const int gc = col0 + r0;
    const unsigned short* bG = ((gc < 512) ? (wqt + gc * 512)
                                           : (wkvt + (gc - 512) * 512)) + c4 * 8;

    auto stage = [&](int buf, int k0) {
        async_copy_b128(aG0 + k0, (uint4*)As[buf] + tid);
        async_copy_b128(aG1 + k0, (uint4*)As[buf] + 256 + tid);
        async_copy_b128(bG + k0, (uint4*)Bt[buf] + tid);
    };

    v8f acc00 = {}, acc01 = {}, acc10 = {}, acc11 = {};

    stage(0, 0);
    for (int it = 0; it < 16; ++it) {
        async_join();                            // this wave's copies landed
        __syncthreads();                         // everyone's copies landed;
                                                 // prior reads of next buf done
        if (it + 1 < 16) {
            stage((it + 1) & 1, (it + 1) * 32);  // overlap copy with compute
            __builtin_prefetch(aG0 + it * 32 + 64, 0, 1);   // L2 warm-up
        }
        const unsigned short* A = As[it & 1];
        const unsigned short* B = Bt[it & 1];
        v16bf a0 = frag_ld(A, wm,      0, 32);
        v16bf a1 = frag_ld(A, wm + 16, 0, 32);
        v16bf b0 = frag_ld(B, wn,      0, 32);
        v16bf b1 = frag_ld(B, wn + 16, 0, 32);
        acc00 = wmma_bf16(a0, b0, acc00);
        acc01 = wmma_bf16(a0, b1, acc01);
        acc10 = wmma_bf16(a1, b0, acc10);
        acc11 = wmma_bf16(a1, b1, acc11);
        // no trailing barrier: s_wait_dscnt precedes WMMA issue, which precedes
        // the next iteration's barrier arrival, so buffer reuse is safe.
    }

    const float qscale = 0.17677669529663687f;   // 1/sqrt(32)
    const int lane = tid & 31;
    const int half = lane >> 4;
    const int nn   = lane & 15;
#pragma unroll
    for (int t = 0; t < 4; ++t) {
        v8f acc = (t == 0) ? acc00 : (t == 1) ? acc01 : (t == 2) ? acc10 : acc11;
        int msub = (t >> 1) * 16;
        int nsub = (t & 1) * 16;
#pragma unroll
        for (int r = 0; r < 8; ++r) {
            int gr = row0 + wm + msub + r + 8 * half;
            int gcc = col0 + wn + nsub + nn;
            float v = acc[r];
            int bwin = gr >> 6;
            int nseq = gr & 63;
            if (gcc < 512) {                     // Q (pre-scaled)
                int h = gcc >> 5, d = gcc & 31;
                Qb[((bwin * 16 + h) * 64 + nseq) * 32 + d] = f2bf((v + bq[gcc]) * qscale);
            } else {
                int c2 = gcc - 512;
                int cc = c2 & 511;
                int h = cc >> 5, d = cc & 31;
                float vb = v + bkv[c2];
                if (c2 < 512) {                  // K: [B,H,N,D]
                    Kb[((bwin * 16 + h) * 64 + nseq) * 32 + d] = f2bf(vb);
                } else {                         // V transposed: [B,H,D,N]
                    Vb[((bwin * 16 + h) * 32 + d) * 64 + nseq] = f2bf(vb);
                }
            }
        }
    }
}

// ---------------------------------------------------------------------------
// Kernel 3: fused attention per (window, head)
// ---------------------------------------------------------------------------
__global__ __launch_bounds__(256)
void attn_kernel(const unsigned short* __restrict__ Qb,
                 const unsigned short* __restrict__ Kb,
                 const unsigned short* __restrict__ Vb,
                 const float* __restrict__ mask,
                 unsigned short* __restrict__ Ob)
{
    __shared__ __attribute__((aligned(16))) unsigned short Qs[64 * 32];
    __shared__ __attribute__((aligned(16))) unsigned short Ks[64 * 32];
    __shared__ __attribute__((aligned(16))) unsigned short Vs[32 * 64];  // [d][m']
    __shared__ __attribute__((aligned(16))) float          Sf[64 * 65];  // padded
    __shared__ __attribute__((aligned(16))) unsigned short Ps[64 * 64];

    const int tid  = threadIdx.x;
    const int wave = tid >> 5;
    const int bh   = blockIdx.x;                 // b*16 + h
    const int b    = bh >> 4;
    const int h    = bh & 15;
    const int w    = b & 255;                    // window-mask index (B/nW = 4)

    {   // stage Q/K/V tiles via async copies: 256 uint4 each, 1 per thread
        const uint4* q4 = (const uint4*)Qb + (size_t)bh * 256 + tid;
        const uint4* k4 = (const uint4*)Kb + (size_t)bh * 256 + tid;
        const uint4* v4 = (const uint4*)Vb + (size_t)bh * 256 + tid;
        async_copy_b128(q4, (uint4*)Qs + tid);
        async_copy_b128(k4, (uint4*)Ks + tid);
        async_copy_b128(v4, (uint4*)Vs + tid);
        async_join();
    }
    __syncthreads();

    const int lane = tid & 31;
    const int half = lane >> 4;
    const int nn   = lane & 15;

    // --- S = (Q*scale) K^T + mask : 16 tiles of 16x16, K = 32
#pragma unroll
    for (int t = 0; t < 2; ++t) {
        int tile = wave * 2 + t;
        int mt = tile >> 2, ntl = tile & 3;
        v16bf a  = frag_ld(Qs, mt * 16, 0, 32);
        v16bf bb = frag_ld(Ks, ntl * 16, 0, 32); // K^T: B-columns = K rows
        v8f c = {};
        c = wmma_bf16(a, bb, c);
#pragma unroll
        for (int r = 0; r < 8; ++r) {
            int m = mt * 16 + r + 8 * half;
            int n = ntl * 16 + nn;
            Sf[m * 65 + n] = c[r] + mask[(w * 64 + m) * 64 + n];
        }
    }
    __syncthreads();

    // --- row softmax (padded stride -> conflict-free)
    if (tid < 64) {
        float* row = &Sf[tid * 65];
        float mx = -1e30f;
#pragma unroll 8
        for (int j = 0; j < 64; ++j) mx = fmaxf(mx, row[j]);
        float sum = 0.f;
#pragma unroll 8
        for (int j = 0; j < 64; ++j) { float e = __expf(row[j] - mx); row[j] = e; sum += e; }
        float inv = 1.0f / sum;
#pragma unroll 8
        for (int j = 0; j < 64; ++j) Ps[tid * 64 + j] = f2bf(row[j] * inv);
    }
    __syncthreads();

    // --- O = P @ V : 8 tiles of 16x16, K = 64
    {
        int mt  = wave & 3;
        int ntl = wave >> 2;                     // 0 or 1 (D = 32)
        v8f c = {};
#pragma unroll
        for (int kk = 0; kk < 64; kk += 32) {
            v16bf a  = frag_ld(Ps, mt * 16, kk, 64);
            v16bf bb = frag_ld(Vs, ntl * 16, kk, 64);   // rows of V^T = out cols
            c = wmma_bf16(a, bb, c);
        }
#pragma unroll
        for (int r = 0; r < 8; ++r) {
            int m = mt * 16 + r + 8 * half;
            int d = ntl * 16 + nn;
            Ob[((size_t)b * 64 + m) * 512 + h * 32 + d] = f2bf(c[r]);
        }
    }
}

// ---------------------------------------------------------------------------
// Kernel 4: output projection  out = O @ Wp + bp  -> fp32 (Wp pre-transposed)
// ---------------------------------------------------------------------------
__global__ __launch_bounds__(256)
void proj_gemm(const unsigned short* __restrict__ Ob,
               const unsigned short* __restrict__ wpt,   // [512 out][512 in]
               const float* __restrict__ bp,
               float* __restrict__ out)
{
    __shared__ __attribute__((aligned(16))) unsigned short As[2][128 * 32];
    __shared__ __attribute__((aligned(16))) unsigned short Bt[2][64 * 32];

    const int tid  = threadIdx.x;
    const int wave = tid >> 5;
    const int row0 = blockIdx.y * 128;
    const int col0 = blockIdx.x * 64;
    const int wm = (wave & 3) * 32;
    const int wn = (wave >> 2) * 32;

    const int r0 = tid >> 2, c4 = tid & 3;
    const unsigned short* aG0 = Ob + (row0 + r0) * 512 + c4 * 8;
    const unsigned short* aG1 = aG0 + 64 * 512;
    const unsigned short* bG  = wpt + (col0 + r0) * 512 + c4 * 8;

    auto stage = [&](int buf, int k0) {
        async_copy_b128(aG0 + k0, (uint4*)As[buf] + tid);
        async_copy_b128(aG1 + k0, (uint4*)As[buf] + 256 + tid);
        async_copy_b128(bG + k0, (uint4*)Bt[buf] + tid);
    };

    v8f acc00 = {}, acc01 = {}, acc10 = {}, acc11 = {};

    stage(0, 0);
    for (int it = 0; it < 16; ++it) {
        async_join();
        __syncthreads();
        if (it + 1 < 16) {
            stage((it + 1) & 1, (it + 1) * 32);
            __builtin_prefetch(aG0 + it * 32 + 64, 0, 1);
        }
        const unsigned short* A = As[it & 1];
        const unsigned short* B = Bt[it & 1];
        v16bf a0 = frag_ld(A, wm,      0, 32);
        v16bf a1 = frag_ld(A, wm + 16, 0, 32);
        v16bf b0 = frag_ld(B, wn,      0, 32);
        v16bf b1 = frag_ld(B, wn + 16, 0, 32);
        acc00 = wmma_bf16(a0, b0, acc00);
        acc01 = wmma_bf16(a0, b1, acc01);
        acc10 = wmma_bf16(a1, b0, acc10);
        acc11 = wmma_bf16(a1, b1, acc11);
    }

    const int lane = tid & 31, half = lane >> 4, nn = lane & 15;
#pragma unroll
    for (int t = 0; t < 4; ++t) {
        v8f acc = (t == 0) ? acc00 : (t == 1) ? acc01 : (t == 2) ? acc10 : acc11;
        int msub = (t >> 1) * 16;
        int nsub = (t & 1) * 16;
#pragma unroll
        for (int r = 0; r < 8; ++r) {
            int gr = row0 + wm + msub + r + 8 * half;
            int gcc = col0 + wn + nsub + nn;
            out[(size_t)gr * 512 + gcc] = acc[r] + bp[gcc];
        }
    }
}

// ---------------------------------------------------------------------------
// Launch
// ---------------------------------------------------------------------------
extern "C" void kernel_launch(void* const* d_in, const int* in_sizes, int n_in,
                              void* d_out, int out_size, void* d_ws, size_t ws_size,
                              hipStream_t stream)
{
    const float* x    = (const float*)d_in[0];   // [1024,64,512]
    const float* mask = (const float*)d_in[1];   // [256,64,64]
    const float* Wq   = (const float*)d_in[2];   // [512,512]
    const float* bq   = (const float*)d_in[3];   // [512]
    const float* Wkv  = (const float*)d_in[4];   // [512,1024]
    const float* bkv  = (const float*)d_in[5];   // [1024]
    const float* Wp   = (const float*)d_in[6];   // [512,512]
    const float* bp   = (const float*)d_in[7];   // [512]
    float* out = (float*)d_out;                  // [1024,64,512] fp32

    char* ws = (char*)d_ws;
    size_t off = 0;
    auto alloc = [&](size_t bytes) -> void* {
        void* p = ws + off;
        off += (bytes + 255) & ~(size_t)255;
        return p;
    };

    const size_t ROWS = 65536;                   // B*N
    unsigned short* xb   = (unsigned short*)alloc(ROWS * 512 * 2);   // 64 MB
    unsigned short* Qb   = (unsigned short*)alloc(ROWS * 512 * 2);
    unsigned short* Kb   = (unsigned short*)alloc(ROWS * 512 * 2);
    unsigned short* Vb   = (unsigned short*)alloc(ROWS * 512 * 2);   // [B,H,D,N]
    unsigned short* WqT  = (unsigned short*)alloc(512 * 512 * 2);
    unsigned short* WkvT = (unsigned short*)alloc(512 * 1024 * 2);
    unsigned short* WpT  = (unsigned short*)alloc(512 * 512 * 2);
    unsigned short* Ob   = xb;                   // x dead after QKV GEMM

    {
        int n = (int)(ROWS * 512);
        cvt_bf16<<<(n + 255) / 256, 256, 0, stream>>>(x, xb, n);
        n = 512 * 512;
        cvtT_bf16<<<(n + 255) / 256, 256, 0, stream>>>(Wq, WqT, 512, 512);
        n = 512 * 1024;
        cvtT_bf16<<<(n + 255) / 256, 256, 0, stream>>>(Wkv, WkvT, 512, 1024);
        n = 512 * 512;
        cvtT_bf16<<<(n + 255) / 256, 256, 0, stream>>>(Wp, WpT, 512, 512);
    }

    qkv_gemm<<<dim3(24, 512), 256, 0, stream>>>(xb, WqT, WkvT, bq, bkv, Qb, Kb, Vb);
    attn_kernel<<<1024 * 16, 256, 0, stream>>>(Qb, Kb, Vb, mask, Ob);
    proj_gemm<<<dim3(8, 512), 256, 0, stream>>>(Ob, WpT, bp, out);
}
